// Antecedent_46222438039892
// MI455X (gfx1250) — matrix operand order repaired and checked by reading, not compiled, not run on hardware
//
#include <hip/hip_runtime.h>
#include <hip/hip_bf16.h>

// Problem constants (from the reference)
#define IN_DIM        8
#define NUM_FS        4
#define NUM_SAM       512
#define NUM_RULE      65536          // 4^8
#define KDIM          32             // NUM_FS * IN_DIM, matches WMMA f16 K exactly

typedef __attribute__((ext_vector_type(16))) _Float16 v16h;
typedef __attribute__((ext_vector_type(8)))  _Float16 v8h;
typedef __attribute__((ext_vector_type(8)))  float    v8f;

// 8 waves per block (wave32), each wave owns one 16-rule (N) tile and sweeps
// all 32 sample (M) tiles. One V_WMMA_F32_16X16X32_F16 per 16x16 output tile.
#define WAVES_PER_BLOCK 8
#define BLOCK_THREADS   (WAVES_PER_BLOCK * 32)
#define N_TILES         (NUM_RULE / 16)    // 4096
#define M_TILES         (NUM_SAM / 16)     // 32
#define GRID_BLOCKS     (N_TILES / WAVES_PER_BLOCK)  // 512

__global__ __launch_bounds__(BLOCK_THREADS)
void Antecedent_46222438039892_kernel(const float* __restrict__ x,      // (512, 8)
                                      const float* __restrict__ center, // (4, 8)
                                      const float* __restrict__ spread, // (4, 8)
                                      const int*   __restrict__ fs_ind, // (65536, 8)
                                      float* __restrict__ out)          // (512, 65536)
{
    // L[s, k] with k = f*8 + d : log membership values, f16, 32 KB in LDS.
    __shared__ _Float16 Lh[NUM_SAM * KDIM];

    const int tid = threadIdx.x;

    // ---------------- Phase 1: build L in LDS cooperatively ----------------
    // Each thread computes 2 sample rows (512 / 256).
    for (int s = tid; s < NUM_SAM; s += BLOCK_THREADS) {
        const float4* xr = (const float4*)(x + s * IN_DIM);
        float4 xa = xr[0], xb = xr[1];
        float xv[IN_DIM] = {xa.x, xa.y, xa.z, xa.w, xb.x, xb.y, xb.z, xb.w};
#pragma unroll
        for (int k = 0; k < KDIM; ++k) {
            const int d = k & 7;
            const float c  = center[k];   // center is (4,8) row-major: idx f*8+d == k
            const float sp = spread[k];
            const float df = xv[d] - c;
            const float val = -(df * df) / (2.0f * sp * sp);
            Lh[s * KDIM + k] = (_Float16)val;
        }
    }
    __syncthreads();

    // ---------------- Phase 2: WMMA over all output tiles ----------------
    const int wave = tid >> 5;
    const int lane = tid & 31;
    const int lane_lo = lane & 15;      // N (rules) / M (samples) within tile
    const int hsel    = lane >> 4;      // 0: low half-lanes, 1: high half-lanes

    const int tile_n = blockIdx.x * WAVES_PER_BLOCK + wave;  // 0..4095, exact
    const int r0 = tile_n * 16;

    // Build B fragment (one-hot of fs_ind) once per wave.
    // ISA B layout (32x16 f16, K x N): N = lane&15; element e -> K = e<8 ? hsel*8+e : 16+hsel*8+(e-8)
    // (matches A fragment k order).
    const int rrow = r0 + lane_lo;
    const int4* fi4 = (const int4*)(fs_ind + (size_t)rrow * IN_DIM);
    const int4 fa = fi4[0], fb = fi4[1];
    const int fr[IN_DIM] = {fa.x, fa.y, fa.z, fa.w, fb.x, fb.y, fb.z, fb.w};
    const int kbase = hsel << 4;
    v16h bfrag;
#pragma unroll
    for (int e = 0; e < 16; ++e) {
        const int k = (e < 8) ? ((hsel << 3) + e) : (8 + (hsel << 3) + e);
        const int f = k >> 3;
        const int d = k & 7;
        bfrag[e] = (fr[d] == f) ? (_Float16)1.0f : (_Float16)0.0f;
    }

    const int col = r0 + lane_lo;

    // Sweep all 32 sample tiles.
    for (int m = 0; m < M_TILES; ++m) {
        const int s0 = m * 16;

        // A fragment (16x32 f16, M x K): lane row M = lane&15;
        // lanes 0-15: element e -> K = e<8 ? e : e+8   (K 0..7, 16..23)
        // lanes 16-31: element e -> K = e<8 ? e+8 : e+16 (K 8..15, 24..31)
        // Both 8-half groups are 16B-contiguous in Lh -> two ds_load_b128.
        const _Float16* abase = &Lh[(s0 + lane_lo) * KDIM + (hsel << 3)];
        const v8h lo = *(const v8h*)(abase);       // K = hsel*8 + 0..7
        const v8h hi = *(const v8h*)(abase + 16);  // K = hsel*8 + 16..23
        const v16h afrag = __builtin_shufflevector(lo, hi,
            0, 1, 2, 3, 4, 5, 6, 7, 8, 9, 10, 11, 12, 13, 14, 15);

        v8f acc = {};
        acc = __builtin_amdgcn_wmma_f32_16x16x32_f16(
            /*neg_a=*/false, afrag, /*neg_b=*/false, bfrag,
            /*c_mod=*/(short)0, acc, /*reuse_a=*/false, /*reuse_b=*/false);

        // C/D layout: VGPR i -> (M = i + hsel*8, N = lane&15).
        const int rowb = s0 + (hsel << 3);
#pragma unroll
        for (int i = 0; i < 8; ++i) {
            const float v = __expf(acc[i]);
            __builtin_nontemporal_store(v, out + (size_t)(rowb + i) * NUM_RULE + col);
        }
    }
}

extern "C" void kernel_launch(void* const* d_in, const int* in_sizes, int n_in,
                              void* d_out, int out_size, void* d_ws, size_t ws_size,
                              hipStream_t stream) {
    const float* x      = (const float*)d_in[0];   // (512, 8)   f32
    const float* center = (const float*)d_in[1];   // (4, 8)     f32
    const float* spread = (const float*)d_in[2];   // (4, 8)     f32
    const int*   fs_ind = (const int*)  d_in[3];   // (65536, 8) i32
    float*       out    = (float*)d_out;           // (512, 65536) f32

    (void)in_sizes; (void)n_in; (void)out_size; (void)d_ws; (void)ws_size;

    Antecedent_46222438039892_kernel<<<GRID_BLOCKS, BLOCK_THREADS, 0, stream>>>(
        x, center, spread, fs_ind, out);
}
